// SRNet_5549097746948
// MI455X (gfx1250) — matrix operand order repaired and checked by reading, not compiled, not run on hardware
//
#include <hip/hip_runtime.h>

// ---------------------------------------------------------------------------
// SRNet forward for MI455X (gfx1250, wave32, WMMA bf16 16x16x32, fp32 accum).
// All matmuls (KNN Gram, EdgeConv MLPs, 1x1 convs) run through
// v_wmma_f32_16x16x32_bf16. EdgeConv is fully fused (async-gather -> edge
// feature in LDS -> GEMM1 -> lrelu -> GEMM2 -> lrelu -> max/sum over k in
// registers). Neighbor gathers and dense-GEMM A tiles use
// GLOBAL_LOAD_ASYNC_TO_LDS_B128 double-buffering (ASYNCcnt) when available.
// ---------------------------------------------------------------------------

#define DEV __device__ __forceinline__

typedef __attribute__((ext_vector_type(16))) __bf16 v16bf;
typedef __attribute__((ext_vector_type(8)))  float  v8f;

union FragBF { v16bf v; unsigned int u[8]; __bf16 h[16]; };

static constexpr int kB  = 8;
static constexpr int kN  = 2048;
static constexpr int kBN = kB * kN;   // 16384
static constexpr int kR  = 8;

// ---- CDNA5 async copy (global -> LDS), guarded ----------------------------
#if defined(__has_builtin)
# if __has_builtin(__builtin_amdgcn_global_load_async_to_lds_b128) && \
     __has_builtin(__builtin_amdgcn_s_wait_asynccnt)
#  define USE_ASYNC 1
# endif
#endif
#ifndef USE_ASYNC
# define USE_ASYNC 0
#endif

#if USE_ASYNC
// builtin signature (from diagnostics): (v4i AS1*, v4i AS3*, imm offset, imm cpol)
typedef int v4i_ __attribute__((vector_size(16)));
typedef v4i_ __attribute__((address_space(1)))* g_v4i_p;
typedef v4i_ __attribute__((address_space(3)))* l_v4i_p;
DEV void async_b128(const float* g, float* l) {
  __builtin_amdgcn_global_load_async_to_lds_b128((g_v4i_p)g, (l_v4i_p)l, 0, 0);
}
# define WAIT_ASYNC(n) __builtin_amdgcn_s_wait_asynccnt(n)
#else
DEV void async_b128(const float* g, float* l) {
  *(float4*)l = *(const float4*)g;
}
# define WAIT_ASYNC(n) do {} while (0)
#endif

DEV v8f wmma_bf16(v16bf a, v16bf b, v8f c) {
  // D = A(16x32) x B(32x16) + C, fp32 accumulate
  return __builtin_amdgcn_wmma_f32_16x16x32_bf16(
      /*neg_a=*/false, a, /*neg_b=*/false, b,
      /*c_mod=*/(short)0, c, /*reuse_a=*/false, /*reuse_b=*/false);
}

// A fragment: 16x32 (MxK) bf16 from row-major buffer (stride in elements).
// ISA 7.12.2: lanes 0-15 M=lane K={0..7,16..23}; lanes 16-31 same M, K+8.
DEV v16bf load_a_frag(const __bf16* src, int stride, int lane) {
  FragBF f;
  const __bf16* row = src + (lane & 15) * stride;
  const int khi = (lane >> 4) * 8;
#pragma unroll
  for (int v = 0; v < 8; ++v) {
    int k = ((v < 4) ? 0 : 16) + khi + 2 * (v & 3);
    f.u[v] = *(const unsigned int*)(row + k);
  }
  return f.v;
}

// Same A layout, but source is fp32 (converted to bf16 during fragment load).
DEV v16bf load_a_frag_f32(const float* src, int stride, int lane) {
  FragBF f;
  const float* row = src + (lane & 15) * stride;
  const int khi = (lane >> 4) * 8;
#pragma unroll
  for (int v = 0; v < 8; ++v) {
    int k = ((v < 4) ? 0 : 16) + khi + 2 * (v & 3);
    f.h[2 * v]     = (__bf16)row[k];
    f.h[2 * v + 1] = (__bf16)row[k + 1];
  }
  return f.v;
}

// B fragment: 32x16 (KxN) bf16. Source is W^T (rows = N, contiguous in K) so
// each K-pair is one b32 load. Lanes 0-15: N=lane, K={0..7,16..23}; lanes 16-31: K+8.
DEV v16bf load_b_frag(const __bf16* wt, int kstride, int lane) {
  FragBF f;
  const __bf16* col = wt + (lane & 15) * kstride + (lane >> 4) * 8;
#pragma unroll
  for (int v = 0; v < 8; ++v) f.u[v] = *(const unsigned int*)(col + ((v < 4) ? 0 : 16) + 2 * (v & 3));
  return f.v;
}

// ---------------------------------------------------------------------------
// Weight convert: fp32 W[K][Co] -> bf16 W^T[CoP][Kp], zero padded.
// ---------------------------------------------------------------------------
__global__ void k_wconv(const float* __restrict__ W, __bf16* __restrict__ Wt,
                        int K, int Co, int Kp) {
  const int n = blockIdx.x;  // 0..CoP-1
  for (int kk = threadIdx.x; kk < Kp; kk += blockDim.x) {
    float v = (n < Co && kk < K) ? W[(size_t)kk * Co + n] : 0.f;
    Wt[(size_t)n * Kp + kk] = (__bf16)v;
  }
}

// ---------------------------------------------------------------------------
// KNN prep: fp32 activations -> bf16 (zero pad to Cp) + squared norms.
// ---------------------------------------------------------------------------
__global__ __launch_bounds__(32)
void k_prep(const float* __restrict__ src, int stride, int C, int Cp,
            __bf16* __restrict__ xb, float* __restrict__ d2) {
  const int row = blockIdx.x;
  const float* r = src + (size_t)row * stride;
  float s = 0.f;
  for (int c = threadIdx.x; c < Cp; c += 32) {
    float v = (c < C) ? r[c] : 0.f;
    xb[(size_t)row * Cp + c] = (__bf16)v;
    s += v * v;
  }
#pragma unroll
  for (int off = 16; off > 0; off >>= 1) s += __shfl_down(s, off);
  if (threadIdx.x == 0) d2[row] = s;
}

// ---------------------------------------------------------------------------
// KNN: Gram via WMMA + per-row top-k (ascending distance, includes self).
// One block = 16 query rows of one batch; query fragments staged in LDS once;
// 4 waves each own a 16-col tile per chunk.
// ---------------------------------------------------------------------------
__global__ __launch_bounds__(128)
void k_knn(const __bf16* __restrict__ xb, const float* __restrict__ d2,
           int Cp, int* __restrict__ idxout, int k) {
  __shared__ __bf16 Ash[16 * 128];
  __shared__ float dch[16][68];
  __shared__ float d2r[16];
  __shared__ float ldist[16][20];
  __shared__ int   lidx[16][20];
  const int tid = threadIdx.x, wave = tid >> 5, lane = tid & 31;
  const int base = blockIdx.x * 16;
  const int bb = base >> 11;          // batch index (N = 2048)
  const int colbase = bb << 11;
  const int nn = lane & 15, mb = (lane >> 4) * 8;

  // stage 16 query rows (bf16) into LDS once
  {
    const unsigned int* asrc = (const unsigned int*)(xb + (size_t)base * Cp);
    unsigned int* adst = (unsigned int*)Ash;
    for (int t = tid; t < 8 * Cp; t += 128) adst[t] = asrc[t];
  }
  if (tid < 16) {
    d2r[tid] = d2[base + tid];
    for (int j = 0; j < k; ++j) { ldist[tid][j] = 3.0e38f; lidx[tid][j] = 0; }
  }
  __syncthreads();

  for (int c0 = 0; c0 < kN; c0 += 64) {
    const int cw = c0 + wave * 16;
    const __bf16* brow = xb + (size_t)(colbase + cw) * Cp;
    v8f acc = {0.f, 0.f, 0.f, 0.f, 0.f, 0.f, 0.f, 0.f};
    for (int k0 = 0; k0 < Cp; k0 += 32) {
      v16bf a = load_a_frag(Ash + k0, Cp, lane);
      v16bf b = load_b_frag(brow + k0, Cp, lane);
      acc = wmma_bf16(a, b, acc);
    }
    const float d2c = d2[colbase + cw + nn];
#pragma unroll
    for (int v = 0; v < 8; ++v)
      dch[mb + v][wave * 16 + nn] = d2r[mb + v] + d2c - 2.0f * acc[v];
    __syncthreads();
    if (tid < 16) {
      const int m = tid;
      for (int cc = 0; cc < 64; ++cc) {
        float dd = dch[m][cc];
        if (dd < ldist[m][k - 1]) {
          int p = k - 1;
          while (p > 0 && ldist[m][p - 1] > dd) {
            ldist[m][p] = ldist[m][p - 1];
            lidx[m][p]  = lidx[m][p - 1];
            --p;
          }
          ldist[m][p] = dd;
          lidx[m][p]  = c0 + cc;
        }
      }
    }
    __syncthreads();
  }
  if (tid < 16)
    for (int j = 0; j < k; ++j)
      idxout[(size_t)(base + tid) * k + j] = colbase + lidx[tid][j];
}

// ---------------------------------------------------------------------------
// Fused EdgeConv (templated on Ci so async-copy wait counts are immediates):
// per 16-point tile, loop neighbors j with double-buffered async gather:
//   e = [xj - xi | xi] (bf16, LDS) -> h1 = lrelu(e W1 + b1)
//   [optional] h2 = lrelu(h1 W2 + b2) -> max/sum over j in registers.
// ---------------------------------------------------------------------------
template <int CI>
__global__ __launch_bounds__(128)
void k_edgeconv(const float* __restrict__ in, int inS,
                const int* __restrict__ idx, int k,
                const __bf16* __restrict__ w1, int Kp1,
                const float* __restrict__ b1, int Cmid,
                const __bf16* __restrict__ w2, const float* __restrict__ b2,
                int Cout, int hasl2, int aggsum,
                const float* __restrict__ res, int resS,
                float* __restrict__ out, int outS) {
  constexpr bool kAsync = (USE_ASYNC != 0) && (CI >= 64);
  constexpr int  kC4    = (CI >= 64) ? (CI / 4) : 1;   // b128 chunks per row
  __shared__ float  xi[16 * 128];
  __shared__ float  Xj[2][16 * (CI >= 64 ? CI : 4)];   // async gather staging
  __shared__ __bf16 Abuf[16 * 256];
  __shared__ __bf16 Hbuf[16 * 256];
  const int tid = threadIdx.x, wave = tid >> 5, lane = tid & 31;
  const int base = blockIdx.x * 16;
  const int nn = lane & 15, mb = (lane >> 4) * 8;

  // issue one neighbor tile's gather (b128 per 4 floats) into Xj[buf]
  auto issue = [&](int j, int buf) {
    for (int t = tid; t < 16 * kC4; t += 128) {
      const int r = t / kC4, c4 = t - r * kC4;
      const int gi = idx[(size_t)(base + r) * k + j];
      async_b128(in + (size_t)gi * inS + c4 * 4, &Xj[buf][r * CI + c4 * 4]);
    }
  };

  for (int t = tid; t < 16 * 256; t += 128) Abuf[t] = (__bf16)0.f;  // pad cols
  for (int t = tid; t < 16 * CI; t += 128) {
    const int r = t / CI, c = t - r * CI;
    xi[r * 128 + c] = in[(size_t)(base + r) * inS + c];
  }
  const int CoutEff = hasl2 ? Cout : Cmid;
  const int otiles  = CoutEff >> 4;
  const int mtiles  = Cmid >> 4;
  v8f agg[4];
#pragma unroll
  for (int i = 0; i < 4; ++i)
#pragma unroll
    for (int v = 0; v < 8; ++v) agg[i][v] = aggsum ? 0.f : -3.0e38f;

  if constexpr (kAsync) issue(0, 0);
  __syncthreads();

  for (int j = 0; j < k; ++j) {
    if constexpr (kAsync) {
      constexpr int kT = CI / 32;   // async instructions per thread per tile
      const int cur = j & 1;
      if (j + 1 < k) { issue(j + 1, cur ^ 1); WAIT_ASYNC(kT); }
      else           { WAIT_ASYNC(0); }
      __syncthreads();              // tile j visible to all waves
      for (int t = tid; t < 16 * CI; t += 128) {
        const int r = t / CI, c = t - r * CI;
        const float xjv = Xj[cur][r * CI + c];
        const float xiv = xi[r * 128 + c];
        Abuf[r * 256 + c]      = (__bf16)(xjv - xiv);
        Abuf[r * 256 + CI + c] = (__bf16)xiv;
      }
    } else {
      for (int t = tid; t < 16 * CI; t += 128) {
        const int r = t / CI, c = t - r * CI;
        const int gi = idx[(size_t)(base + r) * k + j];
        const float xjv = in[(size_t)gi * inS + c];
        const float xiv = xi[r * 128 + c];
        Abuf[r * 256 + c]      = (__bf16)(xjv - xiv);
        Abuf[r * 256 + CI + c] = (__bf16)xiv;
      }
    }
    __syncthreads();

    // GEMM1
    for (int t = wave; t < mtiles; t += 4) {
      v8f c1 = {0.f, 0.f, 0.f, 0.f, 0.f, 0.f, 0.f, 0.f};
      const __bf16* wt = w1 + (size_t)(t * 16) * Kp1;
      __builtin_prefetch(wt, 0, 3);
      for (int k0 = 0; k0 < Kp1; k0 += 32) {
        v16bf a = load_a_frag(Abuf + k0, 256, lane);
        v16bf b = load_b_frag(wt + k0, Kp1, lane);
        c1 = wmma_bf16(a, b, c1);
      }
      const float bv = b1[t * 16 + nn];
      if (hasl2) {
#pragma unroll
        for (int v = 0; v < 8; ++v) {
          float x = c1[v] + bv;
          x = x > 0.f ? x : 0.2f * x;
          Hbuf[(mb + v) * 256 + t * 16 + nn] = (__bf16)x;
        }
      } else {
        const int i = t >> 2;
#pragma unroll
        for (int v = 0; v < 8; ++v) {
          float x = c1[v] + bv;
          x = x > 0.f ? x : 0.2f * x;
          agg[i][v] = aggsum ? (agg[i][v] + x) : fmaxf(agg[i][v], x);
        }
      }
    }
    __syncthreads();

    // GEMM2 + aggregation
    if (hasl2) {
      for (int t = wave; t < otiles; t += 4) {
        v8f c2 = {0.f, 0.f, 0.f, 0.f, 0.f, 0.f, 0.f, 0.f};
        const __bf16* wt2 = w2 + (size_t)(t * 16) * Cmid;
        __builtin_prefetch(wt2, 0, 3);
        for (int k0 = 0; k0 < Cmid; k0 += 32) {
          v16bf a = load_a_frag(Hbuf + k0, 256, lane);
          v16bf b = load_b_frag(wt2 + k0, Cmid, lane);
          c2 = wmma_bf16(a, b, c2);
        }
        const float bv = b2[t * 16 + nn];
        const int i = t >> 2;
#pragma unroll
        for (int v = 0; v < 8; ++v) {
          float x = c2[v] + bv;
          x = x > 0.f ? x : 0.2f * x;
          agg[i][v] = aggsum ? (agg[i][v] + x) : fmaxf(agg[i][v], x);
        }
      }
      __syncthreads();
    }
  }

  // writeout (+ optional residual)
  for (int t = wave; t < otiles; t += 4) {
    const int i = t >> 2;
#pragma unroll
    for (int v = 0; v < 8; ++v) {
      float x = agg[i][v];
      const size_t ro = (size_t)(base + mb + v);
      if (res) x += res[ro * resS + t * 16 + nn];
      out[ro * outS + t * 16 + nn] = x;
    }
  }
}

// ---------------------------------------------------------------------------
// Dense 1x1 conv: out = act(in @ W + b). One wave per 16x16 output tile,
// A tile double-buffered through LDS via async copy (fp32 -> bf16 at frag load).
// ---------------------------------------------------------------------------
__global__ __launch_bounds__(32)
void k_gemm(const float* __restrict__ in, int inS,
            const __bf16* __restrict__ wt, int Kp,
            const float* __restrict__ bias, int Co, int CoS,
            float* __restrict__ out, int act) {
  __shared__ float Atf[2][16 * 32];
  const int lane = threadIdx.x;
  const int row0 = blockIdx.x * 16;
  const int n0 = blockIdx.y * 16;
  const int nn = lane & 15, mb = (lane >> 4) * 8;
  v8f acc = {0.f, 0.f, 0.f, 0.f, 0.f, 0.f, 0.f, 0.f};
  const __bf16* wbase = wt + (size_t)n0 * Kp;

  auto issue = [&](int k0, int buf) {
    for (int t = lane; t < 128; t += 32) {          // 16 rows x 8 b128 chunks
      const int r = t >> 3, c4 = t & 7;
      async_b128(in + (size_t)(row0 + r) * inS + k0 + c4 * 4,
                 &Atf[buf][r * 32 + c4 * 4]);
    }
  };
  issue(0, 0);
  for (int k0 = 0; k0 < Kp; k0 += 32) {
    const int cur = (k0 >> 5) & 1;
    if (k0 + 32 < Kp) { issue(k0 + 32, cur ^ 1); WAIT_ASYNC(4); }
    else              { WAIT_ASYNC(0); }
    __syncthreads();
    v16bf a = load_a_frag_f32(&Atf[cur][0], 32, lane);
    v16bf b = load_b_frag(wbase + k0, Kp, lane);
    acc = wmma_bf16(a, b, acc);
    __syncthreads();
  }
  const float bv = (n0 + nn) < Co ? bias[n0 + nn] : 0.f;
#pragma unroll
  for (int v = 0; v < 8; ++v) {
    float x = acc[v] + bv;
    if (act) x = fmaxf(x, 0.f);
    out[(size_t)(row0 + mb + v) * CoS + n0 + nn] = x;
  }
}

// ---------------------------------------------------------------------------
// Final expansion: expanded = pos_dup + edge * (mask > EPS); also emit mask.
// ---------------------------------------------------------------------------
__global__ __launch_bounds__(128)
void k_expand(const float* __restrict__ pos, const float* __restrict__ edge,
              const float* __restrict__ mask, float* __restrict__ oexp,
              float* __restrict__ omask) {
  const int i = blockIdx.x * blockDim.x + threadIdx.x;
  if (i >= kBN) return;
  const float mv = mask[(size_t)i * 32];
  const float bm = mv > 0.01f ? 1.f : 0.f;
  omask[i] = mv;
  const float px = pos[i * 3 + 0], py = pos[i * 3 + 1], pz = pos[i * 3 + 2];
#pragma unroll
  for (int rr = 0; rr < kR; ++rr) {
    const size_t o = (size_t)(i * kR + rr) * 3;
    oexp[o + 0] = px + edge[(size_t)i * 32 + rr * 3 + 0] * bm;
    oexp[o + 1] = py + edge[(size_t)i * 32 + rr * 3 + 1] * bm;
    oexp[o + 2] = pz + edge[(size_t)i * 32 + rr * 3 + 2] * bm;
  }
}

// ---------------------------------------------------------------------------
// Host orchestration
// ---------------------------------------------------------------------------
extern "C" void kernel_launch(void* const* d_in, const int* in_sizes, int n_in,
                              void* d_out, int out_size, void* d_ws, size_t ws_size,
                              hipStream_t stream) {
  (void)in_sizes; (void)n_in; (void)out_size; (void)ws_size;
  char* wsp = (char*)d_ws;
  size_t off = 0;
  auto alloc = [&](size_t bytes) -> void* {
    void* p = wsp + off;
    off = (off + bytes + 255) & ~(size_t)255;
    return p;
  };

  // activation / scratch buffers
  __bf16* XB  = (__bf16*)alloc((size_t)kBN * 128 * 2);
  float*  D2  = (float*) alloc((size_t)kBN * 4);
  int*    IDX = (int*)   alloc((size_t)kBN * 20 * 4);
  float*  X    = (float*)alloc((size_t)kBN * 128 * 4);
  float*  ENC  = (float*)alloc((size_t)kBN * 256 * 4);
  float*  U64  = (float*)alloc((size_t)kBN * 64 * 4);
  float*  U256 = (float*)alloc((size_t)kBN * 256 * 4);
  float*  S1   = (float*)alloc((size_t)kBN * 32 * 4);
  float*  S2   = (float*)alloc((size_t)kBN * 32 * 4);
  float*  EDGE = (float*)alloc((size_t)kBN * 32 * 4);
  float*  MASK = (float*)alloc((size_t)kBN * 32 * 4);

  auto fin  = [&](int i) { return (const float*)d_in[i]; };
  auto wconv = [&](int i, int K, int Co, int Kp, int CoP) -> __bf16* {
    __bf16* dst = (__bf16*)alloc((size_t)Kp * CoP * 2);
    k_wconv<<<dim3(CoP), dim3(128), 0, stream>>>(fin(i), dst, K, Co, Kp);
    return dst;
  };

  // weights (d_in flattened in sorted-key order: feature, params..., pos)
  __bf16* w_ec0_1 = wconv(1, 6, 128, 32, 128);
  __bf16* w_ec0_2 = wconv(3, 128, 128, 128, 128);
  __bf16* w_id1_1 = wconv(5, 256, 128, 256, 128);
  __bf16* w_id1_2 = wconv(7, 128, 128, 128, 128);
  __bf16* w_id2_1 = wconv(9, 256, 128, 256, 128);
  __bf16* w_id2_2 = wconv(11, 128, 128, 128, 128);
  __bf16* w_mkc1  = wconv(13, 256, 64, 256, 64);
  __bf16* w_mkc2  = wconv(15, 256, 64, 256, 64);
  __bf16* w_mkd0  = wconv(17, 256, 128, 256, 128);
  __bf16* w_mkd1  = wconv(19, 128, 64, 128, 64);
  __bf16* w_mkd2  = wconv(21, 64, 1, 64, 32);
  __bf16* w_mke1_1 = wconv(23, 128, 256, 128, 256);
  __bf16* w_mke1_2 = wconv(25, 256, 256, 256, 256);
  __bf16* w_mke2_1 = wconv(27, 128, 256, 128, 256);
  __bf16* w_upc1  = wconv(29, 256, 64, 256, 64);
  __bf16* w_upc2  = wconv(31, 256, 64, 256, 64);
  __bf16* w_upd0  = wconv(33, 256, 12, 256, 32);
  __bf16* w_upd1  = wconv(35, 12, 24, 32, 32);
  __bf16* w_upd2  = wconv(37, 24, 24, 32, 32);
  __bf16* w_upe1_1 = wconv(39, 128, 256, 128, 256);
  __bf16* w_upe1_2 = wconv(41, 256, 256, 256, 256);
  __bf16* w_upe2_1 = wconv(43, 128, 256, 128, 256);
  __bf16* w_upe2_2 = wconv(45, 256, 256, 256, 256);

  const float* feature = fin(0);
  const float* pos     = fin(47);

  auto prep = [&](const float* src, int stride, int C, int Cp) {
    k_prep<<<dim3(kBN), dim3(32), 0, stream>>>(src, stride, C, Cp, XB, D2);
  };
  auto knn = [&](int Cp, int k) {
    k_knn<<<dim3(kBN / 16), dim3(128), 0, stream>>>(XB, D2, Cp, IDX, k);
  };
  auto econv = [&](int Ci, const float* in, int inS, int k,
                   const __bf16* w1, int Kp1, const float* b1, int Cmid,
                   const __bf16* w2, const float* b2, int Cout,
                   int hasl2, int aggsum, const float* res, int resS,
                   float* out, int outS) {
    if (Ci == 3)
      k_edgeconv<3><<<dim3(kBN / 16), dim3(128), 0, stream>>>(
          in, inS, IDX, k, w1, Kp1, b1, Cmid, w2, b2, Cout,
          hasl2, aggsum, res, resS, out, outS);
    else if (Ci == 64)
      k_edgeconv<64><<<dim3(kBN / 16), dim3(128), 0, stream>>>(
          in, inS, IDX, k, w1, Kp1, b1, Cmid, w2, b2, Cout,
          hasl2, aggsum, res, resS, out, outS);
    else
      k_edgeconv<128><<<dim3(kBN / 16), dim3(128), 0, stream>>>(
          in, inS, IDX, k, w1, Kp1, b1, Cmid, w2, b2, Cout,
          hasl2, aggsum, res, resS, out, outS);
  };
  auto gemm = [&](const float* in, int inS, const __bf16* wt, int Kp,
                  const float* bias, int Co, int CoS, float* out, int act) {
    k_gemm<<<dim3(kBN / 16, CoS / 16), dim3(32), 0, stream>>>(
        in, inS, wt, Kp, bias, Co, CoS, out, act);
  };

  // ---- encoder ----
  prep(feature, 3, 3, 32);  knn(32, 20);
  econv(3, feature, 3, 20, w_ec0_1, 32, fin(2), 128, w_ec0_2, fin(4), 128,
        1, 0, nullptr, 0, X, 128);
  prep(X, 128, 128, 128);   knn(128, 20);
  econv(128, X, 128, 20, w_id1_1, 256, fin(6), 128, w_id1_2, fin(8), 128,
        1, 0, X, 128, ENC, 256);                      // f1 -> ENC[:, :128]
  prep(ENC, 256, 128, 128); knn(128, 20);
  econv(128, ENC, 256, 20, w_id2_1, 256, fin(10), 128, w_id2_2, fin(12), 128,
        1, 0, ENC, 256, ENC + 128, 256);              // f2 -> ENC[:, 128:]

  // ---- upsample branch ----
  gemm(ENC, 256, w_upc1, 256, fin(30), 64, 64, U64, 1);
  prep(U64, 64, 64, 64);    knn(64, 12);
  econv(64, U64, 64, 12, w_upe1_1, 128, fin(40), 256, w_upe1_2, fin(42), 256,
        1, 0, nullptr, 0, U256, 256);
  gemm(U256, 256, w_upc2, 256, fin(32), 64, 64, U64, 1);
  prep(U64, 64, 64, 64);    knn(64, 4);
  econv(64, U64, 64, 4, w_upe2_1, 128, fin(44), 256, w_upe2_2, fin(46), 256,
        1, 0, nullptr, 0, U256, 256);
  gemm(U256, 256, w_upd0, 256, fin(34), 12, 32, S1, 1);
  gemm(S1, 32, w_upd1, 32, fin(36), 24, 32, S2, 1);
  gemm(S2, 32, w_upd2, 32, fin(38), 24, 32, EDGE, 0);

  // ---- mask branch ----
  gemm(ENC, 256, w_mkc1, 256, fin(14), 64, 64, U64, 1);
  prep(U64, 64, 64, 64);    knn(64, 12);
  econv(64, U64, 64, 12, w_mke1_1, 128, fin(24), 256, w_mke1_2, fin(26), 256,
        1, 0, nullptr, 0, U256, 256);
  gemm(U256, 256, w_mkc2, 256, fin(16), 64, 64, U64, 1);
  prep(U64, 64, 64, 64);    knn(64, 8);
  econv(64, U64, 64, 8, w_mke2_1, 128, fin(28), 256, nullptr, nullptr, 256,
        0, 1, nullptr, 0, U256, 256);                 // sum agg, no l2
  gemm(U256, 256, w_mkd0, 256, fin(18), 128, 128, X, 1);
  gemm(X, 128, w_mkd1, 128, fin(20), 64, 64, U64, 1);
  gemm(U64, 64, w_mkd2, 64, fin(22), 1, 32, MASK, 1);

  // ---- final expansion ----
  float* oexp  = (float*)d_out;
  float* omask = oexp + (size_t)kBN * kR * 3;
  k_expand<<<dim3((kBN + 127) / 128), dim3(128), 0, stream>>>(
      pos, EDGE, MASK, oexp, omask);
}
